// Decoder_31662498906685
// MI455X (gfx1250) — compile-verified, matrix operand order checked by reading, not run on hardware
//
#include <hip/hip_runtime.h>
#include <stdint.h>

// Problem constants (match reference: B=1024, T=128, H=1024, POSE=96, TRAJ=32, I=128)
#define BATCH  1024
#define TSTEPS 128
#define HDIM   1024
#define IDIM   128
#define PDIM   96
#define TRJ    32

typedef __attribute__((ext_vector_type(16))) __bf16 bf16x16;
typedef __attribute__((ext_vector_type(8)))  float  floatx8;

// ---------------------------------------------------------------- helpers
__device__ __forceinline__ unsigned short f2bf(float f) {
  unsigned int u = __float_as_uint(f);
  u += 0x7FFFu + ((u >> 16) & 1u);            // round-to-nearest-even
  return (unsigned short)(u >> 16);
}

__device__ __forceinline__ float sigmoidf(float x) {
  return 1.0f / (1.0f + __expf(-x));
}

// Load one 16x32 bf16 WMMA fragment (A or B operand) from a row-major matrix.
// ISA layout (CDNA5 ISA): lane l -> row row0+(l&15); K elems {kb..kb+7, kb+16..kb+23},
// kb = 8*(l>=16). Two 16-byte contiguous loads per lane.
__device__ __forceinline__ bf16x16 load_frag(const unsigned short* __restrict__ base,
                                             int row0, int ld, int k0) {
  const int lane = threadIdx.x & 31;
  const unsigned short* p =
      base + (size_t)(row0 + (lane & 15)) * ld + k0 + ((lane >> 4) << 3);
  union { bf16x16 v; uint4 u[2]; } t;
  t.u[0] = *reinterpret_cast<const uint4*>(p);
  t.u[1] = *reinterpret_cast<const uint4*>(p + 16);
  return t.v;
}

__device__ __forceinline__ floatx8 wmma_bf16(bf16x16 a, bf16x16 b, floatx8 c) {
  return __builtin_amdgcn_wmma_f32_16x16x32_bf16(
      /*neg_a=*/false, a, /*neg_b=*/false, b,
      /*c_mod=*/(short)0, c, /*reuse_a=*/false, /*reuse_b=*/false);
}

// ---------------------------------------------------------------- prep kernels
__global__ void k_cvt(const float* __restrict__ s, unsigned short* __restrict__ d, int n) {
  int i = blockIdx.x * blockDim.x + threadIdx.x;
  if (i < n) d[i] = f2bf(s[i]);
}

__global__ void k_initx(const float* __restrict__ gt,
                        float* __restrict__ xf, unsigned short* __restrict__ xb) {
  int i = blockIdx.x * blockDim.x + threadIdx.x;
  if (i >= BATCH * IDIM) return;
  int b = i / IDIM, c = i % IDIM;
  float v = gt[(size_t)b * TSTEPS * IDIM + c];   // gt[:,0,:]
  xf[i] = v;
  xb[i] = f2bf(v);
}

// ---------------------------------------------------------------- fused GRU cell
// Each wave: 32(batch) x 16(hidden) output tile (2 M sub-tiles) -> the three Whh/Wih
// gate B-fragments are reused across two A-fragments: 5 frag loads per 6 WMMAs.
// grid = (HDIM/64, BATCH/32), block = 128 (4 waves -> 4 column tiles).
__global__ void __launch_bounds__(128) k_gru(
    const unsigned short* __restrict__ hB, const float* __restrict__ hF,
    const unsigned short* __restrict__ xB,
    const unsigned short* __restrict__ WhhB,  // [3H][H] bf16 row-major
    const unsigned short* __restrict__ WihB,  // [3H][I] bf16 row-major
    const float* __restrict__ bih, const float* __restrict__ bhh,
    float* __restrict__ hnF, unsigned short* __restrict__ hnB) {
  const int lane = threadIdx.x & 31;
  const int wave = threadIdx.x >> 5;
  const int n0 = (blockIdx.x * 4 + wave) * 16;  // hidden-column tile
  const int b0 = blockIdx.y * 32;               // batch-row tile (2x16)

  floatx8 aR[2] = {{}, {}}, aZ[2] = {{}, {}}, aXN[2] = {{}, {}}, aHN[2] = {{}, {}};

  // h-part: K = HDIM, gates r,z,n from Whh
#pragma unroll 2
  for (int k = 0; k < HDIM; k += 32) {
    bf16x16 a0 = load_frag(hB, b0,      HDIM, k);
    bf16x16 a1 = load_frag(hB, b0 + 16, HDIM, k);
    bf16x16 br = load_frag(WhhB, n0, HDIM, k);
    bf16x16 bz = load_frag(WhhB + (size_t)HDIM * HDIM, n0, HDIM, k);
    bf16x16 bn = load_frag(WhhB + (size_t)2 * HDIM * HDIM, n0, HDIM, k);
    aR[0]  = wmma_bf16(a0, br, aR[0]);   aR[1]  = wmma_bf16(a1, br, aR[1]);
    aZ[0]  = wmma_bf16(a0, bz, aZ[0]);   aZ[1]  = wmma_bf16(a1, bz, aZ[1]);
    aHN[0] = wmma_bf16(a0, bn, aHN[0]);  aHN[1] = wmma_bf16(a1, bn, aHN[1]);
  }
  // x-part: K = IDIM, gates r,z,n from Wih
#pragma unroll
  for (int k = 0; k < IDIM; k += 32) {
    bf16x16 a0 = load_frag(xB, b0,      IDIM, k);
    bf16x16 a1 = load_frag(xB, b0 + 16, IDIM, k);
    bf16x16 br = load_frag(WihB, n0, IDIM, k);
    bf16x16 bz = load_frag(WihB + (size_t)HDIM * IDIM, n0, IDIM, k);
    bf16x16 bn = load_frag(WihB + (size_t)2 * HDIM * IDIM, n0, IDIM, k);
    aR[0]  = wmma_bf16(a0, br, aR[0]);   aR[1]  = wmma_bf16(a1, br, aR[1]);
    aZ[0]  = wmma_bf16(a0, bz, aZ[0]);   aZ[1]  = wmma_bf16(a1, bz, aZ[1]);
    aXN[0] = wmma_bf16(a0, bn, aXN[0]);  aXN[1] = wmma_bf16(a1, bn, aXN[1]);
  }

  // C/D layout: lane -> n = lane&15 ; vgpr v -> m = v + 8*(lane>=16)
  const int j = n0 + (lane & 15);
  const int mbase = (lane >> 4) << 3;
  const float brj = bih[j] + bhh[j];
  const float bzj = bih[HDIM + j] + bhh[HDIM + j];
  const float bxn = bih[2 * HDIM + j];
  const float bhn = bhh[2 * HDIM + j];
#pragma unroll
  for (int mt = 0; mt < 2; ++mt) {
#pragma unroll
    for (int v = 0; v < 8; ++v) {
      const int b = b0 + mt * 16 + mbase + v;
      const float r  = sigmoidf(aR[mt][v] + brj);
      const float z  = sigmoidf(aZ[mt][v] + bzj);
      const float nn = tanhf(aXN[mt][v] + bxn + r * (aHN[mt][v] + bhn));
      const float hp = hF[(size_t)b * HDIM + j];
      const float hn = (1.0f - z) * nn + z * hp;
      hnF[(size_t)b * HDIM + j] = hn;
      hnB[(size_t)b * HDIM + j] = f2bf(hn);
    }
  }
}

// ---------------------------------------------------------------- pose = h_n @ lp_W^T + lp_b
// Each wave: 32x16 output tile. grid = (PDIM/16, BATCH/32), block = 32
__global__ void __launch_bounds__(32) k_pose(
    const unsigned short* __restrict__ hnB,
    const unsigned short* __restrict__ lpWB,  // [PDIM][HDIM]
    const float* __restrict__ lpb,
    float* __restrict__ poseF, unsigned short* __restrict__ poseB) {
  const int lane = threadIdx.x & 31;
  const int p0 = blockIdx.x * 16;
  const int b0 = blockIdx.y * 32;
  floatx8 acc[2] = {{}, {}};
#pragma unroll 4
  for (int k = 0; k < HDIM; k += 32) {
    bf16x16 a0 = load_frag(hnB, b0,      HDIM, k);
    bf16x16 a1 = load_frag(hnB, b0 + 16, HDIM, k);
    bf16x16 wv = load_frag(lpWB, p0, HDIM, k);
    acc[0] = wmma_bf16(a0, wv, acc[0]);
    acc[1] = wmma_bf16(a1, wv, acc[1]);
  }
  const int p = p0 + (lane & 15);
  const int mbase = (lane >> 4) << 3;
  const float bias = lpb[p];
#pragma unroll
  for (int mt = 0; mt < 2; ++mt) {
#pragma unroll
    for (int v = 0; v < 8; ++v) {
      const int b = b0 + mt * 16 + mbase + v;
      const float val = acc[mt][v] + bias;
      poseF[(size_t)b * PDIM + p] = val;
      poseB[(size_t)b * PDIM + p] = f2bf(val);
    }
  }
}

// ---------------------------------------------------------------- traj + y[:, 0:32]
// traj = [pose | h_n] @ fc_W^T + fc_b  (split concat: K = 96 + 1024)
// y[:,0:32] = x[:,0:32] + traj ; written to out[t] and next-x buffers.
// Each wave: 32x16 output tile. grid = (TRJ/16, BATCH/32), block = 32
__global__ void __launch_bounds__(32) k_traj(
    const unsigned short* __restrict__ poseB, const unsigned short* __restrict__ hnB,
    const unsigned short* __restrict__ fcWB,  // [TRJ][PDIM+HDIM]
    const float* __restrict__ fcb, const float* __restrict__ xF,
    float* __restrict__ out_t, float* __restrict__ xnF, unsigned short* __restrict__ xnB) {
  const int lane = threadIdx.x & 31;
  const int t0 = blockIdx.x * 16;
  const int b0 = blockIdx.y * 32;
  constexpr int FLD = PDIM + HDIM;  // 1120
  floatx8 acc[2] = {{}, {}};
#pragma unroll
  for (int k = 0; k < PDIM; k += 32) {
    bf16x16 a0 = load_frag(poseB, b0,      PDIM, k);
    bf16x16 a1 = load_frag(poseB, b0 + 16, PDIM, k);
    bf16x16 wv = load_frag(fcWB, t0, FLD, k);
    acc[0] = wmma_bf16(a0, wv, acc[0]);
    acc[1] = wmma_bf16(a1, wv, acc[1]);
  }
#pragma unroll 4
  for (int k = 0; k < HDIM; k += 32) {
    bf16x16 a0 = load_frag(hnB, b0,      HDIM, k);
    bf16x16 a1 = load_frag(hnB, b0 + 16, HDIM, k);
    bf16x16 wv = load_frag(fcWB, t0, FLD, PDIM + k);
    acc[0] = wmma_bf16(a0, wv, acc[0]);
    acc[1] = wmma_bf16(a1, wv, acc[1]);
  }
  const int c = t0 + (lane & 15);   // output column in [0, 32)
  const int mbase = (lane >> 4) << 3;
  const float bias = fcb[c];
#pragma unroll
  for (int mt = 0; mt < 2; ++mt) {
#pragma unroll
    for (int v = 0; v < 8; ++v) {
      const int b = b0 + mt * 16 + mbase + v;
      const float y = xF[(size_t)b * IDIM + c] + acc[mt][v] + bias;
      out_t[(size_t)b * (TSTEPS * IDIM) + c] = y;
      xnF[(size_t)b * IDIM + c] = y;
      xnB[(size_t)b * IDIM + c] = f2bf(y);
    }
  }
}

// ---------------------------------------------------------------- y[:, 32:128] = x + pose
__global__ void k_tail(const float* __restrict__ xF, const float* __restrict__ poseF,
                       float* __restrict__ out_t, float* __restrict__ xnF,
                       unsigned short* __restrict__ xnB) {
  int i = blockIdx.x * blockDim.x + threadIdx.x;
  if (i >= BATCH * PDIM) return;
  int b = i / PDIM, c = i % PDIM;
  int col = TRJ + c;
  float y = xF[(size_t)b * IDIM + col] + poseF[i];
  out_t[(size_t)b * (TSTEPS * IDIM) + col] = y;
  xnF[(size_t)b * IDIM + col] = y;
  xnB[(size_t)b * IDIM + col] = f2bf(y);
}

// ---------------------------------------------------------------- host
extern "C" void kernel_launch(void* const* d_in, const int* in_sizes, int n_in,
                              void* d_out, int out_size, void* d_ws, size_t ws_size,
                              hipStream_t stream) {
  (void)in_sizes; (void)n_in; (void)out_size; (void)ws_size;
  const float* h_in = (const float*)d_in[0];
  const float* gt   = (const float*)d_in[1];
  const float* Wih  = (const float*)d_in[2];
  const float* Whh  = (const float*)d_in[3];
  const float* bih  = (const float*)d_in[4];
  const float* bhh  = (const float*)d_in[5];
  const float* lpW  = (const float*)d_in[6];
  const float* lpb  = (const float*)d_in[7];
  const float* fcW  = (const float*)d_in[8];
  const float* fcb  = (const float*)d_in[9];
  float* out = (float*)d_out;

  // workspace carve-up (~22 MB), 256B-aligned slices
  uintptr_t w = (uintptr_t)d_ws;
  auto take = [&](size_t bytes) -> void* {
    w = (w + 255) & ~(uintptr_t)255;
    void* p = (void*)w;
    w += bytes;
    return p;
  };
  unsigned short* WhhB = (unsigned short*)take((size_t)3 * HDIM * HDIM * 2);
  unsigned short* WihB = (unsigned short*)take((size_t)3 * HDIM * IDIM * 2);
  unsigned short* lpWB = (unsigned short*)take((size_t)PDIM * HDIM * 2);
  unsigned short* fcWB = (unsigned short*)take((size_t)TRJ * (PDIM + HDIM) * 2);
  float* hF[2]; unsigned short* hB[2];
  float* xF[2]; unsigned short* xB[2];
  for (int i = 0; i < 2; ++i) {
    hF[i] = (float*)take((size_t)BATCH * HDIM * 4);
    hB[i] = (unsigned short*)take((size_t)BATCH * HDIM * 2);
  }
  for (int i = 0; i < 2; ++i) {
    xF[i] = (float*)take((size_t)BATCH * IDIM * 4);
    xB[i] = (unsigned short*)take((size_t)BATCH * IDIM * 2);
  }
  float* poseF = (float*)take((size_t)BATCH * PDIM * 4);
  unsigned short* poseB = (unsigned short*)take((size_t)BATCH * PDIM * 2);

  // one-time prep (weights -> bf16, state init into parity-1 buffers)
  {
    int n;
    n = 3 * HDIM * HDIM;      k_cvt<<<(n + 255) / 256, 256, 0, stream>>>(Whh, WhhB, n);
    n = 3 * HDIM * IDIM;      k_cvt<<<(n + 255) / 256, 256, 0, stream>>>(Wih, WihB, n);
    n = PDIM * HDIM;          k_cvt<<<(n + 255) / 256, 256, 0, stream>>>(lpW, lpWB, n);
    n = TRJ * (PDIM + HDIM);  k_cvt<<<(n + 255) / 256, 256, 0, stream>>>(fcW, fcWB, n);
    hipMemcpyAsync(hF[1], h_in, (size_t)BATCH * HDIM * 4, hipMemcpyDeviceToDevice, stream);
    n = BATCH * HDIM;         k_cvt<<<(n + 255) / 256, 256, 0, stream>>>(h_in, hB[1], n);
    k_initx<<<(BATCH * IDIM + 255) / 256, 256, 0, stream>>>(gt, xF[1], xB[1]);
  }

  // sequential time loop; state double-buffered (read parity rd, write parity wr)
  for (int t = 0; t < TSTEPS; ++t) {
    const int wr = t & 1, rd = wr ^ 1;
    float* out_t = out + (size_t)t * IDIM;
    k_gru<<<dim3(HDIM / 64, BATCH / 32), 128, 0, stream>>>(
        hB[rd], hF[rd], xB[rd], WhhB, WihB, bih, bhh, hF[wr], hB[wr]);
    k_pose<<<dim3(PDIM / 16, BATCH / 32), 32, 0, stream>>>(
        hB[wr], lpWB, lpb, poseF, poseB);
    k_traj<<<dim3(TRJ / 16, BATCH / 32), 32, 0, stream>>>(
        poseB, hB[wr], fcWB, fcb, xF[rd], out_t, xF[wr], xB[wr]);
    k_tail<<<(BATCH * PDIM + 255) / 256, 256, 0, stream>>>(
        xF[rd], poseF, out_t, xF[wr], xB[wr]);
  }
}
